// attention_33741263077772
// MI455X (gfx1250) — matrix operand order repaired and checked by reading, not compiled, not run on hardware
//
#include <hip/hip_runtime.h>
#include <hip/hip_bf16.h>
#include <math.h>

// ---------------------------------------------------------------------------
// MI455X (gfx1250): all matmuls via v_wmma_f32_16x16x32_bf16 (wave32, f32
// accumulate).  Two GEMM kernels:
//  - k_gemm_bf16: direct-from-global fragments (32x32 tile / wave), used for
//    every irregular shape.
//  - k_gemm_bf16_lds: 128x64 block tile, A/B slabs staged into LDS with
//    global_load_async_to_lds_b128 (ASYNCcnt) double-buffered; used for the
//    four M=768/N=64 pointwise-conv / FC GEMMs that carry most FLOPs.
// ---------------------------------------------------------------------------

typedef __bf16 bf16;
typedef __attribute__((ext_vector_type(16))) __bf16 v16bf;
typedef __attribute__((ext_vector_type(8)))  __bf16 v8bf;
typedef __attribute__((ext_vector_type(8)))  float  v8f;

#define NB_B   128
#define NB_D   768
#define NB_L   64
#define NB_Q   448
#define NB_DL  (NB_D * NB_L)        // 49152
#define NB_C4  (4 * NB_D)           // 3072

static __device__ __forceinline__ float wave_sum(float v) {
  #pragma unroll
  for (int o = 16; o > 0; o >>= 1) v += __shfl_xor(v, o, 32);
  return v;
}
static __device__ __forceinline__ float wave_max(float v) {
  #pragma unroll
  for (int o = 16; o > 0; o >>= 1) v = fmaxf(v, __shfl_xor(v, o, 32));
  return v;
}

static __device__ __forceinline__ v8bf ld8(const bf16* p) {
  return *(const v8bf*)p;          // 16B aligned at every call site
}
static __device__ __forceinline__ v16bf combine(v8bf lo, v8bf hi) {
  return __builtin_shufflevector(lo, hi, 0, 1, 2, 3, 4, 5, 6, 7,
                                         8, 9, 10, 11, 12, 13, 14, 15);
}

// ---------------------------------------------------------------------------
// Direct GEMM: Out[b] = act(alpha*A[b]*B[b]ᵀ + biasM + biasN) + Res
// A: MxK row-major.  B: NxK row-major.  M,N,K multiples of 32 at every call
// site => EXEC all-ones at every WMMA (ISA 7.12 requirement).
// ---------------------------------------------------------------------------
__global__ void k_gemm_bf16(
    const bf16* __restrict__ A, int lda, long long sA,
    const bf16* __restrict__ Bm, int ldb, long long sB,
    float* __restrict__ Out, bf16* __restrict__ OutBf, int ldo, long long sO,
    const float* __restrict__ biasM, long long sBiasM,
    const float* __restrict__ biasN, long long sBiasN,
    const float* __restrict__ Res, long long sR,
    int M, int N, int K, float alpha, int relu, int transOut)
{
  const int batch = blockIdx.y;
  const int wave  = threadIdx.x >> 5;
  const int lane  = threadIdx.x & 31;
  const int tilesN = N >> 5;
  const int tiles  = (M >> 5) * tilesN;
  const int t = blockIdx.x * (blockDim.x >> 5) + wave;
  if (t >= tiles) return;                 // uniform per wave
  const int tm = (t / tilesN) << 5;
  const int tn = (t % tilesN) << 5;

  const int r  = lane & 15;
  const int ka = (lane >> 4) << 3;        // 0 | 8
  const int kb = (lane >> 4) << 3;        // 0 | 8

  const bf16* a0 = A + batch * sA + (long long)(tm + r) * lda + ka;
  const bf16* a1 = a0 + (long long)16 * lda;
  const bf16* b0 = Bm + batch * sB + (long long)(tn + r) * ldb + kb;
  const bf16* b1 = b0 + (long long)16 * ldb;

  v8f acc[2][2];
  #pragma unroll
  for (int i = 0; i < 2; ++i)
    #pragma unroll
    for (int j = 0; j < 2; ++j)
      acc[i][j] = (v8f){0.f, 0.f, 0.f, 0.f, 0.f, 0.f, 0.f, 0.f};

  for (int k0 = 0; k0 < K; k0 += 32) {
    const v16bf A0 = combine(ld8(a0), ld8(a0 + 16));
    const v16bf A1 = combine(ld8(a1), ld8(a1 + 16));
    const v16bf B0 = combine(ld8(b0), ld8(b0 + 16));
    const v16bf B1 = combine(ld8(b1), ld8(b1 + 16));
    a0 += 32; a1 += 32; b0 += 32; b1 += 32;
    acc[0][0] = __builtin_amdgcn_wmma_f32_16x16x32_bf16(
        false, A0, false, B0, (short)0, acc[0][0], false, false);
    acc[0][1] = __builtin_amdgcn_wmma_f32_16x16x32_bf16(
        false, A0, false, B1, (short)0, acc[0][1], false, false);
    acc[1][0] = __builtin_amdgcn_wmma_f32_16x16x32_bf16(
        false, A1, false, B0, (short)0, acc[1][0], false, false);
    acc[1][1] = __builtin_amdgcn_wmma_f32_16x16x32_bf16(
        false, A1, false, B1, (short)0, acc[1][1], false, false);
  }

  #pragma unroll
  for (int i = 0; i < 2; ++i) {
    #pragma unroll
    for (int j = 0; j < 2; ++j) {
      const int n  = tn + j * 16 + r;
      const int mb = tm + i * 16 + ((lane >> 4) << 3);
      #pragma unroll
      for (int rr = 0; rr < 8; ++rr) {
        const int m = mb + rr;
        float v = alpha * acc[i][j][rr];
        if (biasM) v += biasM[batch * sBiasM + m];
        if (biasN) v += biasN[batch * sBiasN + n];
        if (relu)  v = fmaxf(v, 0.f);
        const long long pos = transOut ? ((long long)n * ldo + m)
                                       : ((long long)m * ldo + n);
        if (Res)   v += Res[batch * sR + pos];
        if (Out)   Out[batch * sO + pos]   = v;
        if (OutBf) OutBf[batch * sO + pos] = (bf16)v;
      }
    }
  }
}

// ---------------------------------------------------------------------------
// LDS-staged GEMM for M%128==0, N==64, K%32==0:
//   block tile 128x64 (8 waves = 4x2 of 32x32), A slab 128x32 + B slab 64x32
//   double-buffered in dynamic LDS (24 KB), filled by
//   global_load_async_to_lds_b128 and consumed via ds_load_b128 fragments.
// LDS byte map: A[buf] at buf*8192 (row stride 64B); B[buf] at 16384+buf*4096.
// ---------------------------------------------------------------------------
__global__ void __launch_bounds__(256) k_gemm_bf16_lds(
    const bf16* __restrict__ A, int lda,
    const bf16* __restrict__ Bm, int ldb, long long sB,
    float* __restrict__ Out, int ldo, long long sO,
    const float* __restrict__ biasM,
    const float* __restrict__ Res, long long sR,
    int K, int relu)
{
  extern __shared__ char smem[];
  const int batch = blockIdx.y;
  const int tmB  = blockIdx.x << 7;       // 128 rows of A per block
  const int tid  = threadIdx.x;
  const int wave = tid >> 5, lane = tid & 31;
  const int r  = lane & 15;
  const int ka = (lane >> 4) << 3;        // 0 | 8
  const int kb = (lane >> 4) << 3;        // 0 | 8
  const int wm = (wave >> 1) << 5;        // 0,32,64,96
  const int wn = (wave & 1) << 5;         // 0,32

  const bf16* Bb = Bm + batch * sB;       // A has no batch stride (weights)

  // --- async slab fill: A = 512 16B chunks, B = 256 16B chunks ---
  auto issue = [&](int buf, int k0) {
    #pragma unroll
    for (int cc = 0; cc < 2; ++cc) {
      const int c = tid + (cc << 8);
      const int row = c >> 2, col = c & 3;
      const unsigned long long g = (unsigned long long)(size_t)
          (A + (long long)(tmB + row) * lda + k0 + col * 8);
      const unsigned l = (unsigned)((buf << 13) + (row << 6) + (col << 4));
      asm volatile("global_load_async_to_lds_b128 %0, %1, off"
                   :: "v"(l), "v"(g) : "memory");
    }
    {
      const int row = tid >> 2, col = tid & 3;
      const unsigned long long g = (unsigned long long)(size_t)
          (Bb + (long long)row * ldb + k0 + col * 8);
      const unsigned l = (unsigned)(16384 + (buf << 12) + (row << 6) + (col << 4));
      asm volatile("global_load_async_to_lds_b128 %0, %1, off"
                   :: "v"(l), "v"(g) : "memory");
    }
  };

  v8f acc[2][2];
  #pragma unroll
  for (int i = 0; i < 2; ++i)
    #pragma unroll
    for (int j = 0; j < 2; ++j)
      acc[i][j] = (v8f){0.f, 0.f, 0.f, 0.f, 0.f, 0.f, 0.f, 0.f};

  issue(0, 0);
  const int nk = K >> 5;
  for (int s = 0; s < nk; ++s) {
    asm volatile("s_wait_asynccnt 0" ::: "memory");
    __syncthreads();                       // slab[cur] visible to all waves;
    const int cur = s & 1;                 // everyone done reading slab[cur^1]
    if (s + 1 < nk) issue(cur ^ 1, (s + 1) << 5);

    const bf16* sA = (const bf16*)(smem + (cur << 13));
    const bf16* sB = (const bf16*)(smem + 16384 + (cur << 12));
    const v16bf A0 = combine(ld8(sA + ((wm +  0 + r) << 5) + ka),
                             ld8(sA + ((wm +  0 + r) << 5) + ka + 16));
    const v16bf A1 = combine(ld8(sA + ((wm + 16 + r) << 5) + ka),
                             ld8(sA + ((wm + 16 + r) << 5) + ka + 16));
    const v16bf B0 = combine(ld8(sB + ((wn +  0 + r) << 5) + kb),
                             ld8(sB + ((wn +  0 + r) << 5) + kb + 16));
    const v16bf B1 = combine(ld8(sB + ((wn + 16 + r) << 5) + kb),
                             ld8(sB + ((wn + 16 + r) << 5) + kb + 16));
    acc[0][0] = __builtin_amdgcn_wmma_f32_16x16x32_bf16(
        false, A0, false, B0, (short)0, acc[0][0], false, false);
    acc[0][1] = __builtin_amdgcn_wmma_f32_16x16x32_bf16(
        false, A0, false, B1, (short)0, acc[0][1], false, false);
    acc[1][0] = __builtin_amdgcn_wmma_f32_16x16x32_bf16(
        false, A1, false, B0, (short)0, acc[1][0], false, false);
    acc[1][1] = __builtin_amdgcn_wmma_f32_16x16x32_bf16(
        false, A1, false, B1, (short)0, acc[1][1], false, false);
  }

  #pragma unroll
  for (int i = 0; i < 2; ++i) {
    #pragma unroll
    for (int j = 0; j < 2; ++j) {
      const int n  = wn + j * 16 + r;
      const int mb = tmB + wm + i * 16 + ((lane >> 4) << 3);
      #pragma unroll
      for (int rr = 0; rr < 8; ++rr) {
        const int m = mb + rr;
        float v = acc[i][j][rr] + biasM[m];
        if (relu) v = fmaxf(v, 0.f);
        const long long pos = (long long)m * ldo + n;
        if (Res) v += Res[batch * sR + pos];
        Out[batch * sO + pos] = v;
      }
    }
  }
}

// ---------------------------------------------------------------------------
// Prep: build (C*w3) bf16 (row-major), C^T bf16, Q bf16 (row-major), Q^T bf16,
// c1 = C.w1, c2 = Q.w2.  One 256-thread block per token row.
// ---------------------------------------------------------------------------
__global__ void k_prep(const float* __restrict__ x, const float* __restrict__ W,
                       bf16* __restrict__ CW3bf, bf16* __restrict__ CbfT,
                       bf16* __restrict__ Qbf, bf16* __restrict__ QbfT,
                       float* __restrict__ c1, float* __restrict__ c2)
{
  const int rrow = blockIdx.x;               // 0 .. B*512-1
  const int b = rrow >> 9;
  const int tpos = rrow & 511;
  const bool isC = tpos < NB_L;
  const float* row = x + (long long)rrow * NB_D;
  const float* wv = W + (isC ? 0 : NB_D);
  const float* w3 = W + 2 * NB_D;
  float part = 0.f;
  for (int d = threadIdx.x; d < NB_D; d += blockDim.x) {
    const float v = row[d];
    part += v * wv[d];
    if (isC) {
      CW3bf[((long long)b * NB_L + tpos) * NB_D + d] = (bf16)(v * w3[d]);
      CbfT[((long long)b * NB_D + d) * NB_L + tpos]  = (bf16)v;
    } else {
      const int j = tpos - NB_L;
      Qbf[((long long)b * NB_Q + j) * NB_D + d]  = (bf16)v;
      QbfT[((long long)b * NB_D + d) * NB_Q + j] = (bf16)v;
    }
  }
  part = wave_sum(part);
  __shared__ float red[8];
  const int w = threadIdx.x >> 5, lane = threadIdx.x & 31;
  if (lane == 0) red[w] = part;
  __syncthreads();
  if (threadIdx.x == 0) {
    float s = 0.f;
    for (int i = 0; i < (int)(blockDim.x >> 5); ++i) s += red[i];
    if (isC) c1[b * NB_L + tpos] = s;
    else     c2[b * NB_Q + tpos - NB_L] = s;
  }
}

// Strided softmax, one wave per row, bf16 output.
__global__ void k_softmax(const float* __restrict__ src, bf16* __restrict__ dst,
                          long long sSrc, long long sDst, int nrows, int ncols,
                          long long rs, long long cs, long long drs, long long dcs)
{
  const int row = blockIdx.x * (blockDim.x >> 5) + (threadIdx.x >> 5);
  if (row >= nrows) return;
  const int lane = threadIdx.x & 31;
  const float* p = src + blockIdx.y * sSrc + (long long)row * rs;
  float mx = -3.4e38f;
  for (int c = lane; c < ncols; c += 32) mx = fmaxf(mx, p[(long long)c * cs]);
  mx = wave_max(mx);
  float sum = 0.f;
  for (int c = lane; c < ncols; c += 32) sum += __expf(p[(long long)c * cs] - mx);
  sum = wave_sum(sum);
  const float inv = 1.f / sum;
  bf16* d = dst + blockIdx.y * sDst + (long long)row * drs;
  for (int c = lane; c < ncols; c += 32)
    d[(long long)c * dcs] = (bf16)(__expf(p[(long long)c * cs] - mx) * inv);
}

__global__ void k_cast_bf16(const float* __restrict__ src, bf16* __restrict__ dst,
                            long long n)
{
  const long long i = (long long)blockIdx.x * blockDim.x + threadIdx.x;
  if (i < n) dst[i] = (bf16)src[i];
}

// square transpose-cast: dst[n][k] = src[k][n], MxM
__global__ void k_cast_transpose_sq(const float* __restrict__ src,
                                    bf16* __restrict__ dst, int M)
{
  const long long total = (long long)M * M;
  const long long i = (long long)blockIdx.x * blockDim.x + threadIdx.x;
  if (i >= total) return;
  const int n = (int)(i / M), k = (int)(i % M);
  dst[i] = (bf16)src[(long long)k * M + n];
}

// out = concat([C, A, C*A, C*Bt], dim=2) transposed -> X0 (B, 3072, 64)
__global__ void k_concat(const float* __restrict__ x, const float* __restrict__ Abuf,
                         const float* __restrict__ Bt, float* __restrict__ X0)
{
  const long long total = (long long)NB_B * NB_C4 * NB_L;
  const long long i = (long long)blockIdx.x * blockDim.x + threadIdx.x;
  if (i >= total) return;
  const int b = (int)(i / (NB_C4 * NB_L));
  const int rem = (int)(i % (NB_C4 * NB_L));
  const int c = rem >> 6;
  const int l = rem & 63;
  const int seg = c / NB_D, cc = c % NB_D;
  const float Cv = x[((long long)b * 512 + l) * NB_D + cc];
  float val;
  if (seg == 0) val = Cv;
  else {
    const long long ao = ((long long)b * NB_L + l) * NB_D + cc;
    if (seg == 1)      val = Abuf[ao];
    else if (seg == 2) val = Cv * Abuf[ao];
    else               val = Cv * Bt[ao];
  }
  X0[i] = val;
}

// Depthwise conv over length 64, 'same' padding; bf16 output written
// TRANSPOSED (B, 64, C) so the following GEMM reads it as NxK row-major.
__global__ void k_dwconv(const float* __restrict__ xin, const float* __restrict__ w,
                         const float* __restrict__ bias, bf16* __restrict__ outT,
                         int C, int kw, long long total)
{
  const long long i = (long long)blockIdx.x * blockDim.x + threadIdx.x;
  if (i >= total) return;
  const int l = (int)(i & 63);
  const long long cb = i >> 6;            // b*C + c
  const int c = (int)(cb % C);
  const int b = (int)(cb / C);
  float acc = bias[c];
  const float* p = xin + (cb << 6);
  const int half = kw >> 1;
  for (int t = 0; t < kw; ++t) {
    const int ll = l + t - half;
    if (ll >= 0 && ll < NB_L) acc += p[ll] * w[c * kw + t];
  }
  outT[(((long long)b << 6) + l) * C + c] = (bf16)acc;
}

// X += positional encoding (computed analytically)
__global__ void k_add_pe(float* __restrict__ X)
{
  const long long total = (long long)NB_B * NB_DL;
  const long long i = (long long)blockIdx.x * blockDim.x + threadIdx.x;
  if (i >= total) return;
  const int l = (int)(i & 63);
  const int d = (int)((i >> 6) % NB_D);
  const float fd = (float)d;
  float freq, ph;
  if ((d & 1) == 0) { freq = powf(10000.f, -fd / (float)NB_D); ph = 0.f; }
  else { freq = -powf(10000.f, (1.f - fd) / (float)NB_D); ph = 1.57079632679f; }
  X[i] += sinf((float)l * freq + ph);
}

// LayerNorm over all (D*L) elems of a batch, (D,L) elementwise affine.
// f32 out keeps (D,L); optional bf16 out is TRANSPOSED (L,D).
__global__ void k_ln2(const float* __restrict__ x, const float* __restrict__ w,
                      const float* __restrict__ bws, float* __restrict__ out,
                      bf16* __restrict__ outbfT)
{
  const int b = blockIdx.x;
  const float* p = x + (long long)b * NB_DL;
  float s = 0.f, s2 = 0.f;
  for (int i = threadIdx.x; i < NB_DL; i += blockDim.x) {
    const float v = p[i]; s += v; s2 += v * v;
  }
  s = wave_sum(s); s2 = wave_sum(s2);
  __shared__ float r1[8], r2[8];
  __shared__ float mu_s, rstd_s;
  const int wv = threadIdx.x >> 5, lane = threadIdx.x & 31;
  if (lane == 0) { r1[wv] = s; r2[wv] = s2; }
  __syncthreads();
  if (threadIdx.x == 0) {
    float S = 0.f, S2 = 0.f;
    for (int i = 0; i < (int)(blockDim.x >> 5); ++i) { S += r1[i]; S2 += r2[i]; }
    const float mu = S / (float)NB_DL;
    mu_s = mu;
    rstd_s = rsqrtf(S2 / (float)NB_DL - mu * mu + 1e-5f);
  }
  __syncthreads();
  const float mu = mu_s, rstd = rstd_s;
  float* o = out + (long long)b * NB_DL;
  bf16* ob = outbfT ? outbfT + (long long)b * NB_DL : (bf16*)0;
  for (int i = threadIdx.x; i < NB_DL; i += blockDim.x) {
    const float v = (p[i] - mu) * rstd * w[i] + bws[i];
    o[i] = v;
    if (ob) ob[(long long)(i & 63) * NB_D + (i >> 6)] = (bf16)v;
  }
}

// (B,768,64) f32 -> (B,64,768) bf16
__global__ void k_transpose_cast(const float* __restrict__ X, bf16* __restrict__ xt)
{
  const long long total = (long long)NB_B * NB_DL;
  const long long i = (long long)blockIdx.x * blockDim.x + threadIdx.x;
  if (i >= total) return;
  const int b = (int)(i / NB_DL);
  const int rem = (int)(i % NB_DL);
  const int l = rem / NB_D;
  const int d = rem % NB_D;
  xt[i] = (bf16)X[(long long)b * NB_DL + (long long)d * NB_L + l];
}

// out[b] = sigmoid(dot(X[b], fcf_w) + fcf_b)
__global__ void k_final(const float* __restrict__ X, const float* __restrict__ fw,
                        const float* __restrict__ fb, float* __restrict__ out)
{
  const int b = blockIdx.x;
  const float* p = X + (long long)b * NB_DL;
  float s = 0.f;
  for (int i = threadIdx.x; i < NB_DL; i += blockDim.x) s += p[i] * fw[i];
  s = wave_sum(s);
  __shared__ float red[8];
  const int wv = threadIdx.x >> 5, lane = threadIdx.x & 31;
  if (lane == 0) red[wv] = s;
  __syncthreads();
  if (threadIdx.x == 0) {
    float S = 0.f;
    for (int i = 0; i < 8; ++i) S += red[i];
    out[b] = 1.f / (1.f + __expf(-(S + fb[0])));
  }
}

// ---------------------------------------------------------------------------
// Host side
// ---------------------------------------------------------------------------
static inline int ceil_div(long long a, long long b) { return (int)((a + b - 1) / b); }

static void gemm(hipStream_t s,
                 const bf16* A, int lda, long long sA,
                 const bf16* Bm, int ldb, long long sB,
                 float* Out, bf16* OutBf, int ldo, long long sO,
                 const float* biasM, long long sBm,
                 const float* biasN, long long sBn,
                 const float* Res, long long sR,
                 int M, int N, int K, float alpha, bool relu, bool transOut)
{
  const int tiles = (M / 32) * (N / 32);
  dim3 grid(ceil_div(tiles, 8), NB_B);
  k_gemm_bf16<<<grid, 256, 0, s>>>(A, lda, sA, Bm, ldb, sB,
                                   Out, OutBf, ldo, sO, biasM, sBm, biasN, sBn,
                                   Res, sR, M, N, K, alpha, relu ? 1 : 0,
                                   transOut ? 1 : 0);
}

// LDS/async variant: weights A (no batch stride), N == 64, M % 128 == 0.
static void gemm_lds(hipStream_t s, const bf16* A, int lda,
                     const bf16* Bm, int ldb, long long sB,
                     float* Out, int ldo, long long sO,
                     const float* biasM, const float* Res, long long sR,
                     int M, int K, bool relu)
{
  dim3 grid(M / 128, NB_B);
  k_gemm_bf16_lds<<<grid, 256, 24576, s>>>(A, lda, Bm, ldb, sB,
                                           Out, ldo, sO, biasM, Res, sR,
                                           K, relu ? 1 : 0);
}

extern "C" void kernel_launch(void* const* d_in, const int* in_sizes, int n_in,
                              void* d_out, int out_size, void* d_ws, size_t ws_size,
                              hipStream_t stream)
{
  (void)in_sizes; (void)n_in; (void)out_size; (void)ws_size;
  const float* x       = (const float*)d_in[0];
  const float* W       = (const float*)d_in[1];
  const float* rz_dw_w = (const float*)d_in[2];
  const float* rz_dw_b = (const float*)d_in[3];
  const float* rz_pw_w = (const float*)d_in[4];
  const float* rz_pw_b = (const float*)d_in[5];
  const float* c1_dw_w = (const float*)d_in[6];
  const float* c1_dw_b = (const float*)d_in[7];
  const float* c1_pw_w = (const float*)d_in[8];
  const float* c1_pw_b = (const float*)d_in[9];
  const float* c2_dw_w = (const float*)d_in[10];
  const float* c2_dw_b = (const float*)d_in[11];
  const float* c2_pw_w = (const float*)d_in[12];
  const float* c2_pw_b = (const float*)d_in[13];
  const float* wq      = (const float*)d_in[14];
  const float* wk      = (const float*)d_in[15];
  const float* wv      = (const float*)d_in[16];
  const float* wo      = (const float*)d_in[17];
  const float* fc_w    = (const float*)d_in[18];
  const float* fc_b    = (const float*)d_in[19];
  const float* normb_w = (const float*)d_in[20];
  const float* normb_b = (const float*)d_in[21];
  const float* norm1_w = (const float*)d_in[22];
  const float* norm1_b = (const float*)d_in[23];
  const float* norm2_w = (const float*)d_in[24];
  const float* norm2_b = (const float*)d_in[25];
  const float* norme_w = (const float*)d_in[26];
  const float* norme_b = (const float*)d_in[27];
  const float* fcf_w   = (const float*)d_in[28];
  const float* fcf_b   = (const float*)d_in[29];
  float* out = (float*)d_out;

  // ----- bump allocator over workspace -----
  char* ws = (char*)d_ws;
  size_t off = 0;
  auto alloc = [&](size_t bytes) -> void* {
    void* p = ws + off;
    off += (bytes + 255) & ~(size_t)255;
    return p;
  };

  bf16*  CW3bf  = (bf16*) alloc((size_t)NB_B * NB_L * NB_D * 2);
  bf16*  CbfT   = (bf16*) alloc((size_t)NB_B * NB_L * NB_D * 2);
  bf16*  Qbf    = (bf16*) alloc((size_t)NB_B * NB_Q * NB_D * 2);
  bf16*  QbfT   = (bf16*) alloc((size_t)NB_B * NB_Q * NB_D * 2);
  float* c1v    = (float*)alloc((size_t)NB_B * NB_L * 4);
  float* c2v    = (float*)alloc((size_t)NB_B * NB_Q * 4);
  float* S      = (float*)alloc((size_t)NB_B * NB_L * NB_Q * 4);
  bf16*  S1bf   = (bf16*) alloc((size_t)NB_B * NB_L * NB_Q * 2);
  bf16*  S2bf   = (bf16*) alloc((size_t)NB_B * NB_L * NB_Q * 2);
  float* Abuf   = (float*)alloc((size_t)NB_B * NB_L * NB_D * 4);
  float* T      = (float*)alloc((size_t)NB_B * NB_L * NB_L * 4);
  bf16*  Tbf    = (bf16*) alloc((size_t)NB_B * NB_L * NB_L * 2);
  float* Bt     = (float*)alloc((size_t)NB_B * NB_L * NB_D * 4);
  float* X0     = (float*)alloc((size_t)NB_B * NB_C4 * NB_L * 4);
  bf16*  XdwT   = (bf16*) alloc((size_t)NB_B * NB_C4 * NB_L * 2);
  float* Xa     = (float*)alloc((size_t)NB_B * NB_DL * 4);
  float* Xb     = (float*)alloc((size_t)NB_B * NB_DL * 4);
  float* Xt     = (float*)alloc((size_t)NB_B * NB_DL * 4);
  bf16*  XtTbf  = (bf16*) alloc((size_t)NB_B * NB_DL * 2);
  bf16*  xtbf   = (bf16*) alloc((size_t)NB_B * NB_DL * 2);
  bf16*  qbf    = (bf16*) alloc((size_t)NB_B * NB_DL * 2);
  bf16*  kbf    = (bf16*) alloc((size_t)NB_B * NB_DL * 2);
  bf16*  vTbf   = (bf16*) alloc((size_t)NB_B * NB_DL * 2);
  float* attnS  = (float*)alloc((size_t)NB_B * NB_L * NB_L * 4);
  bf16*  attnbf = (bf16*) alloc((size_t)NB_B * NB_L * NB_L * 2);
  bf16*  sa1bf  = (bf16*) alloc((size_t)NB_B * NB_DL * 2);
  bf16*  dwT    = (bf16*) alloc((size_t)NB_B * NB_DL * 2);
  bf16*  rzpw_bf = (bf16*)alloc((size_t)NB_D * NB_C4 * 2);   // row-major MxK
  bf16*  c1pw_bf = (bf16*)alloc((size_t)NB_D * NB_D * 2);
  bf16*  c2pw_bf = (bf16*)alloc((size_t)NB_D * NB_D * 2);
  bf16*  wqT_bf  = (bf16*)alloc((size_t)NB_D * NB_D * 2);    // transposed NxK
  bf16*  wkT_bf  = (bf16*)alloc((size_t)NB_D * NB_D * 2);
  bf16*  wvT_bf  = (bf16*)alloc((size_t)NB_D * NB_D * 2);
  bf16*  woT_bf  = (bf16*)alloc((size_t)NB_D * NB_D * 2);
  bf16*  fcw_bf  = (bf16*)alloc((size_t)NB_D * NB_D * 2);    // row-major MxK

  // ----- weight casts to bf16 (A operands row-major, B operands transposed) --
  {
    const long long nBig = (long long)NB_D * NB_C4;
    const long long nSq  = (long long)NB_D * NB_D;
    k_cast_bf16<<<ceil_div(nBig, 256), 256, 0, stream>>>(rz_pw_w, rzpw_bf, nBig);
    k_cast_bf16<<<ceil_div(nSq, 256), 256, 0, stream>>>(c1_pw_w, c1pw_bf, nSq);
    k_cast_bf16<<<ceil_div(nSq, 256), 256, 0, stream>>>(c2_pw_w, c2pw_bf, nSq);
    k_cast_bf16<<<ceil_div(nSq, 256), 256, 0, stream>>>(fc_w, fcw_bf, nSq);
    k_cast_transpose_sq<<<ceil_div(nSq, 256), 256, 0, stream>>>(wq, wqT_bf, NB_D);
    k_cast_transpose_sq<<<ceil_div(nSq, 256), 256, 0, stream>>>(wk, wkT_bf, NB_D);
    k_cast_transpose_sq<<<ceil_div(nSq, 256), 256, 0, stream>>>(wv, wvT_bf, NB_D);
    k_cast_transpose_sq<<<ceil_div(nSq, 256), 256, 0, stream>>>(wo, woT_bf, NB_D);
  }

  // ----- phase 1: context-query attention -----
  k_prep<<<NB_B * 512, 256, 0, stream>>>(x, W, CW3bf, CbfT, Qbf, QbfT, c1v, c2v);

  // S = (C*w3) @ Q^T + c1[:,None] + c2[None,:]   (64x448, K=768); B = Q (NxK)
  gemm(stream, CW3bf, NB_D, (long long)NB_L * NB_D,
       Qbf, NB_D, (long long)NB_Q * NB_D,
       S, nullptr, NB_Q, (long long)NB_L * NB_Q,
       c1v, NB_L, c2v, NB_Q, nullptr, 0,
       NB_L, NB_Q, NB_D, 1.f, false, false);

  // S1 = softmax_j(S), S2 = softmax_i(S)
  k_softmax<<<dim3(ceil_div(NB_L, 8), NB_B), 256, 0, stream>>>(
      S, S1bf, (long long)NB_L * NB_Q, (long long)NB_L * NB_Q,
      NB_L, NB_Q, NB_Q, 1, NB_Q, 1);
  k_softmax<<<dim3(ceil_div(NB_Q, 8), NB_B), 256, 0, stream>>>(
      S, S2bf, (long long)NB_L * NB_Q, (long long)NB_L * NB_Q,
      NB_Q, NB_L, 1, NB_Q, 1, NB_Q);

  // A = S1 @ Q  (64x768, K=448); B = Q^T (NxK = 768x448)
  gemm(stream, S1bf, NB_Q, (long long)NB_L * NB_Q,
       QbfT, NB_Q, (long long)NB_Q * NB_D,
       Abuf, nullptr, NB_D, (long long)NB_L * NB_D,
       nullptr, 0, nullptr, 0, nullptr, 0,
       NB_L, NB_D, NB_Q, 1.f, false, false);

  // T = S1 @ S2^T  (64x64, K=448); B = S2 (NxK = 64x448)
  gemm(stream, S1bf, NB_Q, (long long)NB_L * NB_Q,
       S2bf, NB_Q, (long long)NB_L * NB_Q,
       T, nullptr, NB_L, (long long)NB_L * NB_L,
       nullptr, 0, nullptr, 0, nullptr, 0,
       NB_L, NB_L, NB_Q, 1.f, false, false);
  k_cast_bf16<<<ceil_div((long long)NB_B * NB_L * NB_L, 256), 256, 0, stream>>>(
      T, Tbf, (long long)NB_B * NB_L * NB_L);

  // Bt = T @ C  (64x768, K=64); B = C^T (NxK = 768x64)
  gemm(stream, Tbf, NB_L, (long long)NB_L * NB_L,
       CbfT, NB_L, (long long)NB_L * NB_D,
       Bt, nullptr, NB_D, (long long)NB_L * NB_D,
       nullptr, 0, nullptr, 0, nullptr, 0,
       NB_L, NB_D, NB_L, 1.f, false, false);

  // X0 = concat([C, A, C*A, C*Bt]).T  -> (B, 3072, 64)
  k_concat<<<ceil_div((long long)NB_B * NB_C4 * NB_L, 256), 256, 0, stream>>>(
      x, Abuf, Bt, X0);

  // resize front-end: depthwise k=5 then pointwise 3072->768 (LDS/async GEMM)
  k_dwconv<<<ceil_div((long long)NB_B * NB_C4 * NB_L, 256), 256, 0, stream>>>(
      X0, rz_dw_w, rz_dw_b, XdwT, NB_C4, 5, (long long)NB_B * NB_C4 * NB_L);
  gemm_lds(stream, rzpw_bf, NB_C4,
           XdwT, NB_C4, (long long)NB_C4 * NB_L,
           Xa, NB_L, NB_DL, rz_pw_b, nullptr, 0,
           NB_D, NB_C4, false);

  // ----- phase 2: two encoder blocks -----
  float* cur = Xa;
  float* nxt = Xb;
  const float attn_scale = 1.f / sqrtf((float)NB_D);
  const long long nElem = (long long)NB_B * NB_DL;

  for (int blk = 0; blk < 2; ++blk) {
    k_add_pe<<<ceil_div(nElem, 256), 256, 0, stream>>>(cur);

    // conv sublayer 1: res = cur; cur = relu(pw(dw(ln(cur)))) + res
    k_ln2<<<NB_B, 256, 0, stream>>>(cur, normb_w, normb_b, Xt, nullptr);
    k_dwconv<<<ceil_div(nElem, 256), 256, 0, stream>>>(
        Xt, c1_dw_w, c1_dw_b, dwT, NB_D, 3, nElem);
    gemm_lds(stream, c1pw_bf, NB_D, dwT, NB_D, NB_DL,
             nxt, NB_L, NB_DL, c1_pw_b, cur, NB_DL,
             NB_D, NB_D, true);
    { float* t2 = cur; cur = nxt; nxt = t2; }

    // conv sublayer 2
    k_ln2<<<NB_B, 256, 0, stream>>>(cur, norm1_w, norm1_b, Xt, nullptr);
    k_dwconv<<<ceil_div(nElem, 256), 256, 0, stream>>>(
        Xt, c2_dw_w, c2_dw_b, dwT, NB_D, 3, nElem);
    gemm_lds(stream, c2pw_bf, NB_D, dwT, NB_D, NB_DL,
             nxt, NB_L, NB_DL, c2_pw_b, cur, NB_DL,
             NB_D, NB_D, true);
    { float* t2 = cur; cur = nxt; nxt = t2; }

    // self-attention sublayer
    k_ln2<<<NB_B, 256, 0, stream>>>(cur, norm2_w, norm2_b, Xt, nullptr);
    k_transpose_cast<<<ceil_div(nElem, 256), 256, 0, stream>>>(Xt, xtbf);
    // q,k row-major (64x768); v stored transposed (768x64) via transOut
    gemm(stream, xtbf, NB_D, NB_DL, wqT_bf, NB_D, 0,
         nullptr, qbf, NB_D, NB_DL, nullptr, 0, nullptr, 0, nullptr, 0,
         NB_L, NB_D, NB_D, 1.f, false, false);
    gemm(stream, xtbf, NB_D, NB_DL, wkT_bf, NB_D, 0,
         nullptr, kbf, NB_D, NB_DL, nullptr, 0, nullptr, 0, nullptr, 0,
         NB_L, NB_D, NB_D, 1.f, false, false);
    gemm(stream, xtbf, NB_D, NB_DL, wvT_bf, NB_D, 0,
         nullptr, vTbf, NB_L, NB_DL, nullptr, 0, nullptr, 0, nullptr, 0,
         NB_L, NB_D, NB_D, 1.f, false, true);
    // attn = softmax(q @ k^T * scale); B = k (NxK = 64x768)
    gemm(stream, qbf, NB_D, NB_DL, kbf, NB_D, NB_DL,
         attnS, nullptr, NB_L, (long long)NB_L * NB_L,
         nullptr, 0, nullptr, 0, nullptr, 0,
         NB_L, NB_L, NB_D, attn_scale, false, false);
    k_softmax<<<dim3(ceil_div(NB_L, 8), NB_B), 256, 0, stream>>>(
        attnS, attnbf, (long long)NB_L * NB_L, (long long)NB_L * NB_L,
        NB_L, NB_L, NB_L, 1, NB_L, 1);
    // sa1 = attn @ v  (64x768, K=64); B = v^T (NxK = 768x64)
    gemm(stream, attnbf, NB_L, (long long)NB_L * NB_L,
         vTbf, NB_L, NB_DL,
         nullptr, sa1bf, NB_D, NB_DL, nullptr, 0, nullptr, 0, nullptr, 0,
         NB_L, NB_D, NB_L, 1.f, false, false);
    // sa = (sa1 @ wo)^T + res; B = wo^T (NxK = 768x768), transposed store
    gemm(stream, sa1bf, NB_D, NB_DL, woT_bf, NB_D, 0,
         nxt, nullptr, NB_L, NB_DL,
         nullptr, 0, nullptr, 0, cur, NB_DL,
         NB_L, NB_D, NB_D, 1.f, false, true);
    { float* t2 = cur; cur = nxt; nxt = t2; }

    // FC sublayer: cur = relu(fc_w @ ln(cur) + fc_b) + cur  (LDS/async GEMM)
    k_ln2<<<NB_B, 256, 0, stream>>>(cur, norme_w, norme_b, Xt, XtTbf);
    gemm_lds(stream, fcw_bf, NB_D, XtTbf, NB_D, NB_DL,
             nxt, NB_L, NB_DL, fc_b, cur, NB_DL,
             NB_D, NB_D, true);
    { float* t2 = cur; cur = nxt; nxt = t2; }
  }

  // ----- phase 3: final head -----
  k_final<<<NB_B, 256, 0, stream>>>(cur, fcf_w, fcf_b, out);
}
